// DynamicRNNEncoder_32031866094153
// MI455X (gfx1250) — matrix operand order repaired and checked by reading, not compiled, not run on hardware
//
#include <hip/hip_runtime.h>
#include <math.h>

#define B_ 16
#define T_ 1024
#define D_ 128
#define H_ 256

#define UFRAG_BYTES (8 * 16 * 32 * 16 * 2)   // 128 KB: 8 kf x 16 nf x 32 lanes x 16 bf16
#define AGG_BYTES   (B_ * H_ * 4)            // 16 KB
#define RNN_LDS     (UFRAG_BYTES + AGG_BYTES)

typedef __attribute__((ext_vector_type(16))) __bf16 v16bf;
typedef __attribute__((ext_vector_type(8)))  __bf16 v8bf;
typedef __attribute__((ext_vector_type(8)))  float  v8f;
typedef __attribute__((ext_vector_type(4)))  float  v4f;

// GCC-style vector to exactly match the async-LDS builtin's pointee type
typedef int v4i_g __attribute__((vector_size(16)));
typedef __attribute__((address_space(1))) v4i_g gv4i;   // global
typedef __attribute__((address_space(3))) v4i_g lv4i;   // LDS

#if __has_builtin(__builtin_amdgcn_global_load_async_to_lds_b128) && \
    __has_builtin(__builtin_amdgcn_s_wait_asynccnt)
#define USE_ASYNC_LDS 1
#else
#define USE_ASYNC_LDS 0
#endif

// ---------------- pack x (f32) -> bf16, row-major [B*T, D] ----------------
__global__ void pack_x_bf16(const float* __restrict__ x, __bf16* __restrict__ xbf, int n) {
  int i = blockIdx.x * blockDim.x + threadIdx.x;
  if (i < n) xbf[i] = (__bf16)x[i];
}

// ---- pack a row-major f32 [K x N] matrix into WMMA B-fragment bf16 layout ----
// fragment (kf, nf) of 16x16x32 bf16 WMMA:
//   dst[(fi*32 + lane)*16 + e] = src[(kf*32 + (lane&16?8:0) + (e&7) + (e&8?16:0))*N + nf*16 + (lane&15)]
// where fi = kf*(N/16) + nf.  Mapping is bijective: dst[idx] with idx = fi*512+lane*16+e.
__global__ void pack_bfrag(const float* __restrict__ src, __bf16* __restrict__ dst,
                           int K, int N) {
  int idx = blockIdx.x * blockDim.x + threadIdx.x;
  if (idx >= K * N) return;
  int e      = idx & 15;
  int lane   = (idx >> 4) & 31;
  int fi     = idx >> 9;
  int ntiles = N >> 4;
  int nf     = fi % ntiles;
  int kf     = fi / ntiles;
  int k = kf * 32 + ((lane & 16) ? 8 : 0) + (e & 7) + ((e & 8) ? 16 : 0);
  int n = nf * 16 + (lane & 15);
  dst[idx] = (__bf16)src[(size_t)k * N + n];
}

// ---------------- XW = x @ W + bias  (parallel over all B*T rows) ----------------
// grid = (B*T)/16 blocks of 128 threads (4 waves); wave w does n-tiles {4w..4w+3}.
__global__ __launch_bounds__(128)
void xw_gemm(const __bf16* __restrict__ xbf, const __bf16* __restrict__ wp,
             const float* __restrict__ bias, float* __restrict__ xw) {
  const int mi    = blockIdx.x;              // row tile over B*T
  const int lane  = threadIdx.x & 31;
  const int wv    = threadIdx.x >> 5;        // 0..3
  const int row   = mi * 16 + (lane & 15);
  const int khalf = (lane & 16) ? 8 : 0;     // A-frag: hi lanes start at K+8

  v8f acc[4] = {};

  for (int kf = 0; kf < D_ / 32; ++kf) {
    union { v16bf v; v8bf h[2]; } a;
    const __bf16* ap = xbf + (size_t)row * D_ + kf * 32 + khalf;
    a.h[0] = *(const v8bf*)(ap);        // K = base+0..7   (or +8..15 for hi lanes)
    a.h[1] = *(const v8bf*)(ap + 16);   // K = base+16..23 (or +24..31)
#pragma unroll
    for (int t = 0; t < 4; ++t) {
      int nf = wv * 4 + t;
      const v16bf bfrag = *(const v16bf*)(wp + ((size_t)(kf * 16 + nf) * 32 + lane) * 16);
      acc[t] = __builtin_amdgcn_wmma_f32_16x16x32_bf16(
          false, a.v, false, bfrag, (short)0, acc[t], false, false);
    }
  }

  const int rbase = mi * 16 + ((lane & 16) ? 8 : 0);
#pragma unroll
  for (int t = 0; t < 4; ++t) {
    int col = (wv * 4 + t) * 16 + (lane & 15);
    float bv = bias[col];
#pragma unroll
    for (int r = 0; r < 8; ++r)
      xw[(size_t)(rbase + r) * H_ + col] = acc[t][r] + bv;
  }
}

// ---------------- sequential recurrence: one workgroup, 16 waves ----------------
// wave w: (phase 1) b = w  -> sparse aggregation of past h rows (ballot over mask)
//         (phase 2) nf = w -> 16-column tile of agg @ U via 8x wmma bf16 (U in LDS)
__global__ __launch_bounds__(512)
void rnn_seq(const float* __restrict__ cond,   // [B, T, T]
             const __bf16* __restrict__ up,    // packed U fragments (8 kf x 16 nf)
             const float* __restrict__ xw,     // [B*T, H] = x@W + bias
             float* __restrict__ out)          // [B, T, H]
{
  extern __shared__ char smem[];
  const __bf16* uLds = (const __bf16*)smem;          // 128 KB packed U
  float* aggbuf = (float*)(smem + UFRAG_BYTES);      // 16 KB agg exchange

  const int tid  = threadIdx.x;
  const int lane = tid & 31;
  const int wv   = tid >> 5;                   // 0..15

  // ---- stage the loop-invariant 128 KB packed U into LDS (once) ----
#if USE_ASYNC_LDS
  {
    gv4i* g = (gv4i*)up;      // addrspace(1) int4*, const dropped by C-style cast
    lv4i* l = (lv4i*)smem;    // addrspace(3) int4*
#pragma unroll 4
    for (int idx = tid; idx < UFRAG_BYTES / 16; idx += 512)
      __builtin_amdgcn_global_load_async_to_lds_b128(g + idx, l + idx, 0, 0);
    __builtin_amdgcn_s_wait_asynccnt(0);
  }
#else
  for (int idx = tid; idx < UFRAG_BYTES / 16; idx += 512)
    ((v4f*)smem)[idx] = ((const v4f*)up)[idx];
#endif
  __syncthreads();

  const int b    = wv;                         // aggregation batch row
  const int h0   = lane * 8;                   // 8 contiguous h per lane
  const int nf   = wv;                         // output N-tile
  const int arow = lane & 15;                  // A-fragment source row
  const int akh  = (lane & 16) ? 8 : 0;        // A-fragment K sub-offset
  const int crow = (lane & 16) ? 8 : 0;        // C-fragment row offset
  const int ccol = nf * 16 + (lane & 15);      // C-fragment column

  const float* crow0 = cond + (size_t)b * T_ * T_;

  for (int i = 0; i < T_; ++i) {
    // -------- phase 1: agg[b][h0..h0+8) = sum_{t<i, cond!=0} cond * out[b,t,:] --------
    const float* crow_ptr = crow0 + (size_t)i * T_;
    if (i + 1 < T_) __builtin_prefetch(crow_ptr + T_ + lane, 0, 0);  // next step's mask row

    float acc[8];
#pragma unroll
    for (int k = 0; k < 8; ++k) acc[k] = 0.f;

    for (int tb = 0; tb < i; tb += 32) {
      float cl = (tb + lane < i) ? crow_ptr[tb + lane] : 0.f;
      unsigned long long m = __ballot(cl != 0.f);
      while (m) {
        int j = __builtin_ctzll(m);
        m &= m - 1;
        float c = __shfl(cl, j);
        const float* rp = out + ((size_t)b * T_ + (tb + j)) * H_ + h0;
        v4f r0 = *(const v4f*)rp;
        v4f r1 = *(const v4f*)(rp + 4);
        acc[0] += c * r0.x; acc[1] += c * r0.y; acc[2] += c * r0.z; acc[3] += c * r0.w;
        acc[4] += c * r1.x; acc[5] += c * r1.y; acc[6] += c * r1.z; acc[7] += c * r1.w;
      }
    }

    float* ab = aggbuf + b * H_ + h0;
#pragma unroll
    for (int k = 0; k < 8; ++k) ab[k] = acc[k];
    __syncthreads();

    // -------- phase 2: C(16x16) = agg(16x256) @ U(:, nf*16..) via 8 WMMA, U from LDS --------
    v8f cacc = {};
#pragma unroll
    for (int kf = 0; kf < 8; ++kf) {
      union { v16bf v; __bf16 e[16]; } a;
      const float* arp = aggbuf + arow * H_ + kf * 32 + akh;
#pragma unroll
      for (int e2 = 0; e2 < 8; ++e2) {
        a.e[e2]     = (__bf16)arp[e2];        // K = base + 0..7
        a.e[e2 + 8] = (__bf16)arp[e2 + 16];   // K = base + 16..23
      }
      const v16bf bfrag = *(const v16bf*)(uLds + ((size_t)(kf * 16 + nf) * 32 + lane) * 16);
      cacc = __builtin_amdgcn_wmma_f32_16x16x32_bf16(
          false, a.v, false, bfrag, (short)0, cacc, false, false);
    }

    // -------- phase 3: h = tanh(xw + agg@U); publish to history/output --------
#pragma unroll
    for (int r = 0; r < 8; ++r) {
      int bb = r + crow;                              // batch row (M dim)
      size_t o = ((size_t)bb * T_ + i) * H_ + ccol;
      out[o] = tanhf(xw[o] + cacc[r]);
    }
    __threadfence();   // make h_i globally visible before step i+1's gathers
    __syncthreads();
  }
}

extern "C" void kernel_launch(void* const* d_in, const int* in_sizes, int n_in,
                              void* d_out, int out_size, void* d_ws, size_t ws_size,
                              hipStream_t stream) {
  const float* x    = (const float*)d_in[0];   // [B, T, D]
  const float* cond = (const float*)d_in[1];   // [B, T, T]
  const float* W    = (const float*)d_in[2];   // [D, H]
  const float* U    = (const float*)d_in[3];   // [H, H]
  const float* bias = (const float*)d_in[4];   // [H]
  float* out = (float*)d_out;                  // [B, T, H]

  // workspace layout
  char* ws = (char*)d_ws;
  float*  xw  = (float*)ws;                                        // 16 MB
  __bf16* xbf = (__bf16*)(ws + (size_t)B_ * T_ * H_ * 4);          // 4 MB
  __bf16* wp  = (__bf16*)((char*)xbf + (size_t)B_ * T_ * D_ * 2);  // 64 KB
  __bf16* up  = (__bf16*)((char*)wp + (size_t)D_ * H_ * 2);        // 128 KB

  // allow 144 KB dynamic LDS for the sequential kernel (CDNA5: up to 320 KB/WG)
  (void)hipFuncSetAttribute(reinterpret_cast<const void*>(rnn_seq),
                            hipFuncAttributeMaxDynamicSharedMemorySize, RNN_LDS);

  pack_x_bf16<<<(B_ * T_ * D_ + 255) / 256, 256, 0, stream>>>(x, xbf, B_ * T_ * D_);
  pack_bfrag<<<(D_ * H_ + 255) / 256, 256, 0, stream>>>(W, wp, D_, H_);
  pack_bfrag<<<(H_ * H_ + 255) / 256, 256, 0, stream>>>(U, up, H_, H_);
  xw_gemm<<<B_ * T_ / 16, 128, 0, stream>>>(xbf, wp, bias, xw);
  rnn_seq<<<1, 512, RNN_LDS, stream>>>(cond, up, xw, out);
}
